// GeometricEmbedding_15857019257414
// MI455X (gfx1250) — matrix operand order repaired and checked by reading, not compiled, not run on hardware
//
#include <hip/hip_runtime.h>
#include <hip/hip_bf16.h>
#include <float.h>

// Problem constants (match reference setup_inputs)
#define BATCH 4
#define SEQ   256
#define HID   64
#define KP    8

typedef __attribute__((ext_vector_type(16))) _Float16 v16h;
typedef __attribute__((ext_vector_type(8)))  float    v8f;

// div[j] = 10000^(-j/32) = 10^(-0.125*j), j = 0..31  (sinusoidal frequencies)
static constexpr float DIVTAB[32] = {
    1.0f,            0.7498942093f,   0.5623413252f,   0.4216965034f,
    0.3162277660f,   0.2371373706f,   0.1778279410f,   0.1333521432f,
    0.1f,            0.0749894209f,   0.0562341325f,   0.0421696503f,
    0.0316227766f,   0.0237137371f,   0.0177827941f,   0.0133352143f,
    0.01f,           0.0074989421f,   0.0056234133f,   0.0042169650f,
    0.0031622777f,   0.0023713737f,   0.0017782794f,   0.0013335214f,
    0.001f,          0.00074989421f,  0.00056234133f,  0.00042169650f,
    0.00031622777f,  0.00023713737f,  0.00017782794f,  0.00013335214f
};

// ---------------------------------------------------------------------------
// Kernel 1: per (b,i) find the 8 nearest neighbors (incl. self) and store the
// reference vectors c[nbr]-c[i] to workspace: refv[(b*SEQ+i)*KP + s][3].
// One wave32 per (b,i); each lane owns 8 candidate j's; 8 rounds of wave-wide
// (min,idx) shuffle reduction implement top-k selection.
// ---------------------------------------------------------------------------
__global__ __launch_bounds__(32)
void knn_kernel(const float* __restrict__ cent, float* __restrict__ refv) {
    const int bi   = blockIdx.x;          // 0 .. BATCH*SEQ-1
    const int b    = bi >> 8;
    const int i    = bi & (SEQ - 1);
    const int lane = threadIdx.x;

    const float* cb = cent + (size_t)b * SEQ * 3;
    const float cix = cb[i * 3 + 0];
    const float ciy = cb[i * 3 + 1];
    const float ciz = cb[i * 3 + 2];

    float dist[8];
#pragma unroll
    for (int t = 0; t < 8; ++t) {
        const int j = lane + 32 * t;
        const float dx = cb[j * 3 + 0] - cix;
        const float dy = cb[j * 3 + 1] - ciy;
        const float dz = cb[j * 3 + 2] - ciz;
        dist[t] = dx * dx + dy * dy + dz * dz;
    }

    for (int s = 0; s < KP; ++s) {
        // local argmin over this lane's 8 candidates
        float bv = dist[0];
        int   bt = 0;
#pragma unroll
        for (int t = 1; t < 8; ++t)
            if (dist[t] < bv) { bv = dist[t]; bt = t; }
        float v  = bv;
        int   vj = lane + 32 * bt;
        // wave32 (min, idx) reduction; ties -> smaller index (argsort-stable)
#pragma unroll
        for (int off = 16; off > 0; off >>= 1) {
            const float ov = __shfl_xor(v, off, 32);
            const int   oj = __shfl_xor(vj, off, 32);
            if (ov < v || (ov == v && oj < vj)) { v = ov; vj = oj; }
        }
        // owner lane retires the winner
        if ((vj & 31) == lane) dist[vj >> 5] = FLT_MAX;
        // one lane records ref vector for neighbor s
        if (lane == 0) {
            float* o = refv + ((size_t)bi * KP + s) * 3;
            o[0] = cb[vj * 3 + 0] - cix;
            o[1] = cb[vj * 3 + 1] - ciy;
            o[2] = cb[vj * 3 + 2] - ciz;
        }
    }
}

// ---------------------------------------------------------------------------
// WMMA fragment helpers (gfx1250 16-bit layouts, wave32).
//
// A 16x32 f16 fragment: lane L holds row M = L&15; element e maps to
//   K = 16*(e>>3) + 8*(L>>4) + (e&7)          (CDNA5 ISA §7.12.2)
// B 32x16 f16 fragment: lane L holds col N = L&15; element e maps to
//   K = 16*(e>>3) + 8*(L>>4) + (e&7)                          (same order as A)
// C/D 16x16 f32: VGPR r, lanes 0-15: (M=r, N=L); lanes 16-31: (M=8+r, N=L-16)
// ---------------------------------------------------------------------------

// Load one 32(K)x16(N) B-tile of W^T, i.e. B[K][N] = W[N][K]; W is 64x64 f32
// row-major. n = N-tile (0..3), h = K-half (0..1). Two runs of 8 contiguous f32 per lane (K = 8*(L>>4) + (e&7) + 16*(e>>3)).
__device__ __forceinline__ v16h load_wtile(const float* __restrict__ W,
                                           int n, int h, int lane) {
    const int row = n * 16 + (lane & 15);               // N (output channel)
    const int col = h * 32 + ((lane >> 4) << 3);        // K base
    const float* p = W + row * HID + col;
    v16h r;
#pragma unroll
    for (int e = 0; e < 16; ++e) r[e] = (_Float16)p[(e & 7) + ((e >> 3) << 4)];
    return r;
}

// Build one 16(M)x32(K) A-tile of sinusoidal embeddings in-register.
// xr: 16 per-row scalars in LDS. h = K-half (0..1).
// emb[K] = sin(x*div[K/2]) for even K, cos(...) for odd K; div[j]=10^(-j/8).
// Lane's +8 K-offset (hi lanes) is a constant frequency ratio 10^(-0.5).
__device__ __forceinline__ v16h make_emb_tile(const float* xr, int h, int lane) {
    const float x     = xr[lane & 15];
    const float scale = (lane & 16) ? 0.3162277660168379f : 1.0f; // 10^-0.5
    v16h r;
#pragma unroll
    for (int e = 0; e < 16; ++e) {
        const int jbase = 8 * (e >> 3) + ((e & 7) >> 1) + 16 * h; // compile-time
        const float om  = x * (DIVTAB[jbase] * scale);
        const float v   = (e & 1) ? __cosf(om) : __sinf(om);
        r[e] = (_Float16)v;
    }
    return r;
}

__device__ __forceinline__ v8f wmma_f16(v16h a, v16h b, v8f c) {
    return __builtin_amdgcn_wmma_f32_16x16x32_f16(
        /*neg_a=*/false, a, /*neg_b=*/false, b,
        /*c_mod=*/(short)0, c, /*reuse_a=*/false, /*reuse_b=*/false);
}

// ---------------------------------------------------------------------------
// Kernel 2: one wave32 per (b, i, 16-wide j tile).
//  - 16 dists + 128 angles -> LDS
//  - a-branch: 8 (k) x 4 (N-tile) x 2 (K-half) WMMA, running max over k
//  - d-branch: 4 x 2 WMMA; out = d + max_a + (bd + ba)
// ---------------------------------------------------------------------------
__global__ __launch_bounds__(32)
void emb_kernel(const float* __restrict__ cent, const float* __restrict__ refv,
                const float* __restrict__ Wd, const float* __restrict__ bd,
                const float* __restrict__ Wa, const float* __restrict__ ba,
                float* __restrict__ out) {
    __shared__ float s_ref[KP * 3];     // ref vectors for row i
    __shared__ float s_ang[KP][16];     // angle[k][j-row]
    __shared__ float s_dst[16];         // dist[j-row]

    const int blk  = blockIdx.x;                 // b*SEQ*16 + i*16 + jt
    const int jt   = blk & 15;
    const int i    = (blk >> 4) & (SEQ - 1);
    const int b    = blk >> 12;
    const int lane = threadIdx.x;
    const int j0   = jt * 16;

    const float* cb = cent + (size_t)b * SEQ * 3;
    const float cix = cb[i * 3 + 0];
    const float ciy = cb[i * 3 + 1];
    const float ciz = cb[i * 3 + 2];

    if (lane < KP * 3) s_ref[lane] = refv[((size_t)(b * SEQ + i)) * (KP * 3) + lane];
    __syncthreads();

    // anchor vector for this lane's j-row
    const int jr = lane & 15;
    const int j  = j0 + jr;
    const float ax = cb[j * 3 + 0] - cix;
    const float ay = cb[j * 3 + 1] - ciy;
    const float az = cb[j * 3 + 2] - ciz;
    if (lane < 16) s_dst[jr] = ax * ax + ay * ay + az * az;

    // 128 angles: each lane computes 4 (j-row fixed, 4 of the 8 k's)
#pragma unroll
    for (int kk = 0; kk < 4; ++kk) {
        const int k  = ((lane >> 4) << 2) + kk;
        const float rx = s_ref[k * 3 + 0];
        const float ry = s_ref[k * 3 + 1];
        const float rz = s_ref[k * 3 + 2];
        const float cx = ry * az - rz * ay;
        const float cy = rz * ax - rx * az;
        const float cz = rx * ay - ry * ax;
        const float sv = sqrtf(cx * cx + cy * cy + cz * cz);
        const float cv = rx * ax + ry * ay + rz * az;
        s_ang[k][jr] = atan2f(sv, cv);
    }
    __syncthreads();

    // Wa B-tiles resident in VGPRs: 4 N-tiles x 2 K-halves = 64 VGPRs
    v16h WaT[4][2];
#pragma unroll
    for (int n = 0; n < 4; ++n) {
        WaT[n][0] = load_wtile(Wa, n, 0, lane);
        WaT[n][1] = load_wtile(Wa, n, 1, lane);
    }

    // a-branch: project each of the 8 angle embeddings, max over k
    v8f amax[4];
#pragma unroll
    for (int n = 0; n < 4; ++n)
#pragma unroll
        for (int r = 0; r < 8; ++r) amax[n][r] = -FLT_MAX;

    for (int k = 0; k < KP; ++k) {                 // runtime loop: code size
        const v16h A0 = make_emb_tile(&s_ang[k][0], 0, lane);
        const v16h A1 = make_emb_tile(&s_ang[k][0], 1, lane);
#pragma unroll
        for (int n = 0; n < 4; ++n) {
            v8f c = {};
            c = wmma_f16(A0, WaT[n][0], c);
            c = wmma_f16(A1, WaT[n][1], c);
#pragma unroll
            for (int r = 0; r < 8; ++r) amax[n][r] = fmaxf(amax[n][r], c[r]);
        }
    }

    // d-branch + combine + store
    const v16h D0 = make_emb_tile(s_dst, 0, lane);
    const v16h D1 = make_emb_tile(s_dst, 1, lane);
    const size_t rowbase = ((size_t)(b * SEQ + i)) * SEQ; // row of j=0
#pragma unroll
    for (int n = 0; n < 4; ++n) {
        const v16h W0 = load_wtile(Wd, n, 0, lane);
        const v16h W1 = load_wtile(Wd, n, 1, lane);
        v8f c = {};
        c = wmma_f16(D0, W0, c);
        c = wmma_f16(D1, W1, c);
        const int   o    = n * 16 + (lane & 15);
        const float bias = bd[o] + ba[o];          // max(x+ba) == max(x)+ba
#pragma unroll
        for (int r = 0; r < 8; ++r) {
            const int m  = r + ((lane >> 4) << 3); // M of this C/D component
            const int jj = j0 + m;
            out[(rowbase + jj) * HID + o] = c[r] + amax[n][r] + bias;
        }
    }
}

extern "C" void kernel_launch(void* const* d_in, const int* in_sizes, int n_in,
                              void* d_out, int out_size, void* d_ws, size_t ws_size,
                              hipStream_t stream) {
    const float* cent = (const float*)d_in[0];  // (B,S,3)
    const float* Wd   = (const float*)d_in[1];  // (64,64)
    const float* bd   = (const float*)d_in[2];  // (64)
    const float* Wa   = (const float*)d_in[3];  // (64,64)
    const float* ba   = (const float*)d_in[4];  // (64)
    float*       out  = (float*)d_out;          // (B,S,S,64)
    float*       refv = (float*)d_ws;           // (B,S,KP,3) scratch

    knn_kernel<<<BATCH * SEQ, 32, 0, stream>>>(cent, refv);
    emb_kernel<<<BATCH * SEQ * (SEQ / 16), 32, 0, stream>>>(
        cent, refv, Wd, bd, Wa, ba, out);
}
